// EdgeBlock_40827959116111
// MI455X (gfx1250) — compile-verified, matrix operand order checked by reading, not run on hardware
//
#include <hip/hip_runtime.h>
#include <hip/hip_bf16.h>

// ---------------------------------------------------------------------------
// EdgeBlock: per-edge MLP  out[e] = ReLU([x[src]|x[dst]] @ W1 + b1) @ W2 + b2
//   x:   [50000, 128] f32      edge_index: [2, E] int (E = 600000)
//   W1:  [256, 128]  b1: [128] W2: [128, 128]  b2: [128]
//   out: [E, 128] f32
//
// MI455X (gfx1250, wave32) strategy:
//   - 16 edges per wave -> A[16x256] f16 tile; v_wmma_f32_16x16x32_f16
//   - weights fp32->f16, pre-packed in LDS in exact B-fragment order
//   - output tile staged in LDS (reused gather buffer) -> coalesced 512B row
//     stores with non-temporal hint (write-once stream, keep x in L2)
// ---------------------------------------------------------------------------

typedef __attribute__((ext_vector_type(16))) _Float16 v16h;
typedef __attribute__((ext_vector_type(4)))  _Float16 v4h;
typedef __attribute__((ext_vector_type(8)))  float    v8f;
typedef __attribute__((ext_vector_type(4)))  float    v4f;

#define D        128
#define TWOD     256
#define NWAVES   8
#define BLOCK    (NWAVES * 32)

// LDS layout (halves / floats)
#define W1P_HALFS   (8 * 8 * 32 * 16)     // 32768 halves = 64 KB  (kk=8, nt=8)
#define W2P_HALFS   (4 * 8 * 32 * 16)     // 16384 halves = 32 KB  (kk=4, nt=8)
#define TILE_HALFS  (16 * 256 + 16 * 128) // gather/out-stage 8KB + h 4KB
#define SMEM_BYTES  ((W1P_HALFS + W2P_HALFS) * 2 + 256 * 4 + NWAVES * TILE_HALFS * 2)

// Build an A fragment (16x32 f16, ISA layout) from an LDS row-major tile.
// lane 0-15: M=lane, K-halves {kb+0..7, kb+16..23} with kb=0
// lane 16-31: M=lane-16, kb=8
__device__ __forceinline__ v16h load_a_frag(const _Float16* buf, int rowStride,
                                            int lane, int kcol0) {
    const int m  = lane & 15;
    const int kb = (lane >> 4) << 3;
    const _Float16* row = buf + m * rowStride + kcol0 + kb;
    union { v16h v; _Float16 h[16]; } u;
#pragma unroll
    for (int vv = 0; vv < 4; ++vv) {
        u.h[2 * vv]         = row[2 * vv];
        u.h[2 * vv + 1]     = row[2 * vv + 1];
        u.h[8 + 2 * vv]     = row[16 + 2 * vv];
        u.h[8 + 2 * vv + 1] = row[16 + 2 * vv + 1];
    }
    return u.v;
}

// B fragment: pre-packed contiguous 16 halves per lane per (kk,nt) block.
__device__ __forceinline__ v16h load_b_frag(const _Float16* pack, int blk, int lane) {
    return *reinterpret_cast<const v16h*>(pack + ((size_t)blk * 32 + lane) * 16);
}

__global__ void __launch_bounds__(BLOCK)
edgeblock_wmma_kernel(const float* __restrict__ x,
                      const int*   __restrict__ eidx,   // [2*E]
                      const float* __restrict__ W1g,    // [256*128]
                      const float* __restrict__ b1g,    // [128]
                      const float* __restrict__ W2g,    // [128*128]
                      const float* __restrict__ b2g,    // [128]
                      float*       __restrict__ out,    // [E*128]
                      int E, int numTiles, int totalWaves) {
    extern __shared__ char smem[];
    _Float16* w1p   = reinterpret_cast<_Float16*>(smem);            // 32768 halves
    _Float16* w2p   = w1p + W1P_HALFS;                              // 16384 halves
    float*    ldsB1 = reinterpret_cast<float*>(w2p + W2P_HALFS);    // 128 f32
    float*    ldsB2 = ldsB1 + 128;                                  // 128 f32
    _Float16* tiles = reinterpret_cast<_Float16*>(ldsB2 + 128);     // 8*6144 halves

    const int tid  = threadIdx.x;
    const int lane = tid & 31;
    const int wid  = tid >> 5;

    // ---- pack W1/W2 fp32 -> f16 into WMMA B-fragment order -----------------
    // B block (kk,nt): lane L covers column N = nt*16 + L%16,
    // K-halves {kb+0..7, kb+16..23}, kb = (L/16)*8, relative to kk*32.
    for (int idx = tid; idx < W1P_HALFS; idx += BLOCK) {
        const int i   = idx & 15;
        const int ln  = (idx >> 4) & 31;
        const int blk = idx >> 9;          // kk*8 + nt
        const int nt  = blk & 7;
        const int kk  = blk >> 3;
        const int kb  = (ln >> 4) << 3;
        const int vv  = i >> 1;
        const int w   = i & 1;
        const int krel = (vv < 4) ? (kb + 2 * vv + w) : (kb + 16 + 2 * (vv - 4) + w);
        const int K = kk * 32 + krel;
        const int N = nt * 16 + (ln & 15);
        w1p[idx] = (_Float16)W1g[K * D + N];
    }
    for (int idx = tid; idx < W2P_HALFS; idx += BLOCK) {
        const int i   = idx & 15;
        const int ln  = (idx >> 4) & 31;
        const int blk = idx >> 9;
        const int nt  = blk & 7;
        const int kk  = blk >> 3;
        const int kb  = (ln >> 4) << 3;
        const int vv  = i >> 1;
        const int w   = i & 1;
        const int krel = (vv < 4) ? (kb + 2 * vv + w) : (kb + 16 + 2 * (vv - 4) + w);
        const int K = kk * 32 + krel;
        const int N = nt * 16 + (ln & 15);
        w2p[idx] = (_Float16)W2g[K * D + N];
    }
    if (tid < D) { ldsB1[tid] = b1g[tid]; ldsB2[tid] = b2g[tid]; }
    __syncthreads();

    _Float16* gbuf   = tiles + (size_t)wid * TILE_HALFS;  // 16 x 256 f16 gather
    _Float16* hbuf   = gbuf + 16 * 256;                   // 16 x 128 f16 h
    float*    ostage = reinterpret_cast<float*>(gbuf);    // 16 x 128 f32 (reuse)

    const int hi8   = (lane >> 4) << 3;   // 0 or 8 (C-layout row offset)
    const int col16 = lane & 15;          // C-layout column within n-tile
    const int gwave = blockIdx.x * NWAVES + wid;

    for (int t = gwave; t < numTiles; t += totalWaves) {
        // ---- gather 16 edges: [src|dst] rows -> LDS as f16 ----------------
        const int eI  = t * 16 + (lane & 15);
        int nodeIdx = 0;
        if (eI < E) {
            nodeIdx = (lane < 16) ? __builtin_nontemporal_load(eidx + eI)
                                  : __builtin_nontemporal_load(eidx + E + eI);
        }

#pragma unroll 4
        for (int r = 0; r < 16; ++r) {
            const int si = __shfl(nodeIdx, r, 32);
            const int di = __shfl(nodeIdx, r + 16, 32);
            const float4 sv = reinterpret_cast<const float4*>(x + (size_t)si * D)[lane];
            const float4 dv = reinterpret_cast<const float4*>(x + (size_t)di * D)[lane];
            v4h sh, dh;
            sh[0] = (_Float16)sv.x; sh[1] = (_Float16)sv.y;
            sh[2] = (_Float16)sv.z; sh[3] = (_Float16)sv.w;
            dh[0] = (_Float16)dv.x; dh[1] = (_Float16)dv.y;
            dh[2] = (_Float16)dv.z; dh[3] = (_Float16)dv.w;
            *reinterpret_cast<v4h*>(gbuf + r * TWOD + lane * 4)       = sh;
            *reinterpret_cast<v4h*>(gbuf + r * TWOD + D + lane * 4)   = dh;
        }

        // ---- GEMM1: h = relu(A[16x256] @ W1 + b1), h -> LDS (f16) ----------
#pragma unroll
        for (int nt = 0; nt < 8; ++nt) {
            v8f acc = {};
#pragma unroll
            for (int kk = 0; kk < 8; ++kk) {
                const v16h a = load_a_frag(gbuf, TWOD, lane, kk * 32);
                const v16h b = load_b_frag(w1p, kk * 8 + nt, lane);
                acc = __builtin_amdgcn_wmma_f32_16x16x32_f16(
                          false, a, false, b, (short)0, acc, false, false);
            }
            const float bias = ldsB1[nt * 16 + col16];
#pragma unroll
            for (int r = 0; r < 8; ++r) {
                float f = acc[r] + bias;
                f = fmaxf(f, 0.0f);
                hbuf[(r + hi8) * D + nt * 16 + col16] = (_Float16)f;
            }
        }

        // ---- GEMM2: out = h[16x128] @ W2 + b2 -> LDS stage -----------------
        // (gather buffer is dead now; reuse it as a 16x128 f32 staging tile)
#pragma unroll
        for (int nt = 0; nt < 8; ++nt) {
            v8f acc = {};
#pragma unroll
            for (int kk = 0; kk < 4; ++kk) {
                const v16h a = load_a_frag(hbuf, D, lane, kk * 32);
                const v16h b = load_b_frag(w2p, kk * 8 + nt, lane);
                acc = __builtin_amdgcn_wmma_f32_16x16x32_f16(
                          false, a, false, b, (short)0, acc, false, false);
            }
            const float bias = ldsB2[nt * 16 + col16];
            const int   col  = nt * 16 + col16;
#pragma unroll
            for (int r = 0; r < 8; ++r) {
                ostage[(r + hi8) * D + col] = acc[r] + bias;
            }
        }

        // ---- flush: 16 coalesced 512B row stores, non-temporal -------------
#pragma unroll 4
        for (int r = 0; r < 16; ++r) {
            const int row = t * 16 + r;
            if (row < E) {
                const v4f v = *reinterpret_cast<const v4f*>(ostage + r * D + lane * 4);
                __builtin_nontemporal_store(
                    v, reinterpret_cast<v4f*>(out + (size_t)row * D) + lane);
            }
        }
    }
}

extern "C" void kernel_launch(void* const* d_in, const int* in_sizes, int n_in,
                              void* d_out, int out_size, void* d_ws, size_t ws_size,
                              hipStream_t stream) {
    const float* x    = (const float*)d_in[0];
    const int*   eidx = (const int*)d_in[1];
    const float* W1   = (const float*)d_in[2];
    const float* b1   = (const float*)d_in[3];
    const float* W2   = (const float*)d_in[4];
    const float* b2   = (const float*)d_in[5];
    float*       out  = (float*)d_out;

    const int E        = in_sizes[1] / 2;
    const int numTiles = (E + 15) / 16;

    const int grid = 512;
    const int totalWaves = grid * NWAVES;

    // Allow >64KB dynamic LDS (CDNA5: up to 320 KB per workgroup).
    (void)hipFuncSetAttribute((const void*)edgeblock_wmma_kernel,
                              hipFuncAttributeMaxDynamicSharedMemorySize,
                              SMEM_BYTES);

    edgeblock_wmma_kernel<<<grid, BLOCK, SMEM_BYTES, stream>>>(
        x, eidx, W1, b1, W2, b2, out, E, numTiles, totalWaves);
}
